// Encoder_17360257810780
// MI455X (gfx1250) — matrix operand order repaired and checked by reading, not compiled, not run on hardware
//
#include <hip/hip_runtime.h>

typedef _Float16 half8  __attribute__((ext_vector_type(8)));
typedef _Float16 half16 __attribute__((ext_vector_type(16)));
typedef float    float8 __attribute__((ext_vector_type(8)));

#define WMMA_F16(a, b, c) \
    __builtin_amdgcn_wmma_f32_16x16x32_f16(false, (a), false, (b), (short)0, (c), false, false)

// Build A-fragment for v_wmma_f32_16x16x32_f16.
// Lane L holds row M = L%16; lanes 0-15 hold K{c0..c0+7, c0+16..c0+23} with c0=0,
// lanes 16-31 with c0=8 (ISA 7.12.2, 16-bit A 16x32 layout).
// Caller passes p = &row[kstep*32 + c0]; elements 0..7 <- p[0..7], 8..15 <- p[16..23].
__device__ __forceinline__ half16 make_afrag(const _Float16* p) {
    half8 lo = *(const half8*)p;
    half8 hi = *(const half8*)(p + 16);
    half16 r;
#pragma unroll
    for (int j = 0; j < 8; ++j) { r[j] = lo[j]; r[j + 8] = hi[j]; }
    return r;
}

// ---------------------------------------------------------------------------
// Pack a column-major-consumed [Ktot x Ntot] f32 matrix into the dense 16-bit
// B 32x16 WMMA lane layout: dst[((ntile*KSTEPS + ks)*32 + lane)*16 + j]
//   N = ntile*16 + lane%16,  K = ks*32 + (j<8?0:16) + (lane<16?0:8) + j%8
// Rows k >= Kvalid are zero-padded (for conv1's K=45 -> 64).
// ---------------------------------------------------------------------------
__global__ __launch_bounds__(256) void pack_b_f16(const float* __restrict__ Wm,
                                                  _Float16* __restrict__ dst,
                                                  int Ktot, int Ntot, int Kvalid) {
    int total = (Ntot >> 4) * (Ktot >> 5) * 512;
    int e = blockIdx.x * 256 + threadIdx.x;
    if (e >= total) return;
    int j    = e & 15;
    int lane = (e >> 4) & 31;
    int rest = e >> 9;
    int ksteps = Ktot >> 5;
    int ks = rest % ksteps;
    int nt = rest / ksteps;
    int n = nt * 16 + (lane & 15);
    int k = ks * 32 + ((j & 8) << 1) + ((lane >> 4) << 3) + (j & 7);
    dst[e] = (k < Kvalid) ? (_Float16)Wm[(size_t)k * Ntot + n] : (_Float16)0.f;
}

// ---------------------------------------------------------------------------
// conv1 3x3 s2 SAME 5->32 via WMMA implicit GEMM, +bias (folded into C),
// relu, emit f16 NHWC [512][32][32][32]. One block (256 thr = 8 waves) per
// frame; frame staged in LDS as f16 with 1-px zero halo: [66][66][5].
// K = 45 taps*ch zero-padded to 64 -> 2 K-steps; N = 32 ch -> 2 tiles;
// M-tile = 16 consecutive output pixels. A-fragment element addresses are
// base(position) + delta(kappa), delta a compile-time constant per element
// (lane-half selects between two constants with one v_cndmask).
// ---------------------------------------------------------------------------
__global__ __launch_bounds__(256) void conv1_wmma(const float* __restrict__ x,
                                                  const _Float16* __restrict__ wpack,
                                                  const float* __restrict__ cb1,
                                                  _Float16* __restrict__ a1) {
    __shared__ _Float16 xt[66 * 66 * 5];   // [iy+1][ix+1][ci] halo-padded
    int tid = threadIdx.x;
    int f   = blockIdx.x;
    // zero (halo), then stage frame f32 -> f16
    {
        uint32_t* p = (uint32_t*)xt;
        for (int i = tid; i < 66 * 66 * 5 / 2; i += 256) p[i] = 0u;
    }
    __syncthreads();
    {
        const float* src = x + (size_t)f * 20480;         // 64*64*5
        for (int e = tid; e < 20480; e += 256) {
            int iy = e / 320;                              // 320 = 64px*5ch per row
            int rem = e - iy * 320;
            xt[(iy + 1) * 330 + 5 + rem] = (_Float16)src[e];
        }
    }
    __syncthreads();
    int wave = tid >> 5, lane = tid & 31;
    int p = lane & 15;
    bool lolane = (lane < 16);
    // invariant B fragments: 2 N-tiles x 2 K-steps (32 VGPRs)
    half16 bf[2][2];
#pragma unroll
    for (int nt = 0; nt < 2; ++nt)
#pragma unroll
        for (int s = 0; s < 2; ++s)
            bf[nt][s] = *(const half16*)(wpack + ((nt * 2 + s) * 32 + lane) * 16);
    float bias0 = cb1[lane & 15];
    float bias1 = cb1[16 + (lane & 15)];

    for (int m = wave; m < 64; m += 8) {                  // 64 M-tiles per frame
        int oy = m >> 1, ox0 = (m & 1) << 4;
        int base = 10 * (oy * 66 + ox0 + p);              // (2*oy*66 + 2*ox)*5
        float8 acc0, acc1;
#pragma unroll
        for (int v = 0; v < 8; ++v) { acc0[v] = bias0; acc1[v] = bias1; }
#pragma unroll
        for (int s = 0; s < 2; ++s) {
            half16 af;
#pragma unroll
            for (int j = 0; j < 16; ++j) {
                int jj = (j < 8) ? j : j + 8;
                int ka = s * 32 + jj;                      // kappa for lanes 0-15
                int kb = ka + 8;                           // kappa for lanes 16-31
                bool va = ka < 45, vb = kb < 45;
                if (!va && !vb) { af[j] = (_Float16)0.f; continue; }
                int da = va ? (330 * (ka / 15) + (ka % 15))
                            : (330 * (kb / 15) + (kb % 15));
                int db = vb ? (330 * (kb / 15) + (kb % 15)) : da;
                int off = base + (lolane ? da : db);
                _Float16 v16 = xt[off];
                if (!(va && vb)) {                         // one lane-half invalid
                    bool keep = va ? lolane : !lolane;
                    v16 = keep ? v16 : (_Float16)0.f;
                }
                af[j] = v16;
            }
            acc0 = WMMA_F16(af, bf[0][s], acc0);
            acc1 = WMMA_F16(af, bf[1][s], acc1);
        }
        // relu + store (C/D element (M = v+8*(lane/16), N))
        _Float16* outp = a1 + ((size_t)f * 1024 + m * 16) * 32;
#pragma unroll
        for (int v = 0; v < 8; ++v) {
            int M = v + 8 * (lane >> 4);
            float t0 = acc0[v] > 0.f ? acc0[v] : 0.f;
            float t1 = acc1[v] > 0.f ? acc1[v] : 0.f;
            outp[M * 32 + (lane & 15)]      = (_Float16)t0;
            outp[M * 32 + 16 + (lane & 15)] = (_Float16)t1;
        }
    }
}

// ---------------------------------------------------------------------------
// conv2 3x3 s2 SAME 32->64 via WMMA implicit GEMM + relu + global-avg-pool +
// dense(64->256)+relu, fused per frame. One block (256 thr = 8 waves) per frame.
// LDS stages 17 padded input rows per half-frame chunk (37 KB).
// ---------------------------------------------------------------------------
__global__ __launch_bounds__(256) void conv2_pool_dense(const _Float16* __restrict__ a1,
                                                        const _Float16* __restrict__ wpack,
                                                        const float* __restrict__ cb2,
                                                        const float* __restrict__ dw,
                                                        const float* __restrict__ db,
                                                        _Float16* __restrict__ feats) {
    __shared__ _Float16 act[17 * 34 * 32];   // [row][col(pad)][ci]
    __shared__ float pool[64];
    int tid  = threadIdx.x;
    int f    = blockIdx.x;
    int wave = tid >> 5, lane = tid & 31;
    int ox = lane & 15;
    int c0 = (lane >> 4) * 8;
    if (tid < 64) pool[tid] = 0.f;

    for (int chunk = 0; chunk < 2; ++chunk) {
        __syncthreads();
        {
            uint32_t* p = (uint32_t*)act;
            for (int i = tid; i < 17 * 34 * 32 / 2; i += 256) p[i] = 0u;
        }
        __syncthreads();
        {
            for (int idx = tid; idx < 17 * 64; idx += 256) {
                int r   = idx >> 6;       // LDS row 0..16
                int c16 = idx & 63;       // 64 uint4 per row
                int iy  = 16 * chunk - 1 + r;
                if (iy >= 0 && iy < 32) {
                    const uint4* src = (const uint4*)(a1 + (size_t)f * 32768 + iy * 1024);
                    uint4* dst = (uint4*)(act + (r * 34 + 1) * 32) + c16;
                    *dst = src[c16];
                }
            }
        }
        __syncthreads();
        float8 acc[4] = {};
        for (int ky = 0; ky < 3; ++ky) {
            for (int kx = 0; kx < 3; ++kx) {
                int tap = ky * 3 + kx;
                const _Float16* ap =
                    act + ((2 * wave + ky) * 34 + (2 * ox + kx)) * 32 + c0;
                half16 afrag = make_afrag(ap);
#pragma unroll
                for (int cg = 0; cg < 4; ++cg) {
                    half16 bfrag =
                        *(const half16*)(wpack + ((tap * 4 + cg) * 32 + lane) * 16);
                    acc[cg] = WMMA_F16(afrag, bfrag, acc[cg]);
                }
            }
        }
#pragma unroll
        for (int cg = 0; cg < 4; ++cg) {
            int n = cg * 16 + (lane & 15);
            float b = cb2[n];
            float s = 0.f;
#pragma unroll
            for (int v = 0; v < 8; ++v) {
                float t = acc[cg][v] + b;
                s += (t > 0.f) ? t : 0.f;
            }
            s += __shfl_xor(s, 16);
            if (lane < 16) atomicAdd(&pool[n], s);
        }
    }
    __syncthreads();
    {
        int d = tid;
        float a = db[d];
        for (int k = 0; k < 64; ++k)
            a = fmaf(pool[k] * (1.f / 256.f), dw[k * 256 + d], a);
        feats[(size_t)f * 256 + d] = (_Float16)(a > 0.f ? a : 0.f);
    }
}

// ---------------------------------------------------------------------------
// xg = feats @ wx + b_gru : [512x256]@[256x768], f16 WMMA, f32 out.
// ---------------------------------------------------------------------------
__global__ __launch_bounds__(256) void xg_gemm(const _Float16* __restrict__ feats,
                                               const _Float16* __restrict__ wxpack,
                                               const float* __restrict__ bg,
                                               float* __restrict__ xg) {
    int wave = threadIdx.x >> 5, lane = threadIdx.x & 31;
    int tile = blockIdx.x * 8 + wave;      // 0..1535
    int mt = tile / 48, nt = tile % 48;
    int row = mt * 16 + (lane & 15);
    int c0 = (lane >> 4) * 8;
    float8 acc = {};
#pragma unroll
    for (int ks = 0; ks < 8; ++ks) {
        half16 afrag = make_afrag(feats + (size_t)row * 256 + ks * 32 + c0);
        half16 bfrag = *(const half16*)(wxpack + ((nt * 8 + ks) * 32 + lane) * 16);
        acc = WMMA_F16(afrag, bfrag, acc);
    }
    int col = nt * 16 + (lane & 15);
    float b = bg[col];
#pragma unroll
    for (int v = 0; v < 8; ++v) {
        int r = mt * 16 + v + 8 * (lane >> 4);
        xg[(size_t)r * 768 + col] = acc[v] + b;
    }
}

// ---------------------------------------------------------------------------
// GRU: 32 sequential steps, persistent single block (512 thr = 16 waves).
// ---------------------------------------------------------------------------
__global__ __launch_bounds__(512) void gru_seq(const float* __restrict__ xg,
                                               const _Float16* __restrict__ whpack,
                                               float* __restrict__ out) {
    __shared__ _Float16 h16[16 * 256];
    __shared__ float    h32[16 * 256];
    __shared__ float    zbuf[16 * 256];
    __shared__ _Float16 rh16[16 * 256];
    int tid  = threadIdx.x;
    int wave = tid >> 5, lane = tid & 31;
    int ln = lane & 15;
    int c0 = (lane >> 4) * 8;
    for (int i = tid; i < 4096; i += 512) { h32[i] = 0.f; h16[i] = (_Float16)0.f; }
    __syncthreads();

    for (int step = 0; step < 32; ++step) {
#pragma unroll
        for (int half_ = 0; half_ < 2; ++half_) {
            int t = wave + 16 * half_;
            float8 acc = {};
#pragma unroll
            for (int ks = 0; ks < 8; ++ks) {
                half16 afrag = make_afrag(h16 + ln * 256 + ks * 32 + c0);
                half16 bfrag = *(const half16*)(whpack + ((t * 8 + ks) * 32 + lane) * 16);
                acc = WMMA_F16(afrag, bfrag, acc);
            }
            int colg = t * 16 + ln;
#pragma unroll
            for (int v = 0; v < 8; ++v) {
                int b = v + 8 * (lane >> 4);
                float g = acc[v] + xg[(size_t)(b * 32 + step) * 768 + colg];
                float s = 1.f / (1.f + __expf(-g));
                if (half_ == 0) zbuf[b * 256 + colg] = s;
                else {
                    int d = colg - 256;
                    rh16[b * 256 + d] = (_Float16)(s * h32[b * 256 + d]);
                }
            }
        }
        __syncthreads();
        {
            int t = 32 + wave;
            int d0 = wave * 16 + ln;
            float8 acc = {};
#pragma unroll
            for (int ks = 0; ks < 8; ++ks) {
                half16 afrag = make_afrag(rh16 + ln * 256 + ks * 32 + c0);
                half16 bfrag = *(const half16*)(whpack + ((t * 8 + ks) * 32 + lane) * 16);
                acc = WMMA_F16(afrag, bfrag, acc);
            }
#pragma unroll
            for (int v = 0; v < 8; ++v) {
                int b = v + 8 * (lane >> 4);
                float hc = tanhf(acc[v] + xg[(size_t)(b * 32 + step) * 768 + 512 + d0]);
                float z  = zbuf[b * 256 + d0];
                float hp = h32[b * 256 + d0];
                float hn = z * hp + (1.f - z) * hc;
                out[(size_t)(b * 32 + step) * 256 + d0] = hn;
                h32[b * 256 + d0] = hn;
                h16[b * 256 + d0] = (_Float16)hn;
            }
        }
        __syncthreads();
    }
    for (int i = tid; i < 4096; i += 512) out[131072 + i] = h32[i];
}

// ---------------------------------------------------------------------------
extern "C" void kernel_launch(void* const* d_in, const int* in_sizes, int n_in,
                              void* d_out, int out_size, void* d_ws, size_t ws_size,
                              hipStream_t stream) {
    const float* x   = (const float*)d_in[0];
    const float* cw1 = (const float*)d_in[1];
    const float* cb1 = (const float*)d_in[2];
    const float* cw2 = (const float*)d_in[3];
    const float* cb2 = (const float*)d_in[4];
    const float* dw  = (const float*)d_in[5];
    const float* db  = (const float*)d_in[6];
    const float* wx  = (const float*)d_in[7];
    const float* wh  = (const float*)d_in[8];
    const float* bg  = (const float*)d_in[9];
    float* out = (float*)d_out;

    char* ws = (char*)d_ws;
    _Float16* a1    = (_Float16*)(ws + 0);          // 512*1024*32 f16 = 33.5 MB
    _Float16* feats = (_Float16*)(ws + 33554432);   // 512*256 f16
    float*    xg    = (float*)   (ws + 33816576);   // 512*768 f32
    _Float16* pc2   = (_Float16*)(ws + 35389440);   // 9*4*512 f16
    _Float16* pwx   = (_Float16*)(ws + 35426304);   // 48*8*512 f16
    _Float16* pwh   = (_Float16*)(ws + 35819520);   // 48*8*512 f16
    _Float16* pc1   = (_Float16*)(ws + 36212736);   // 2*2*512 f16 (conv1, K 45->64)

    // weight repacks into WMMA B-fragment layout
    pack_b_f16<<<8, 256, 0, stream>>>(cw1, pc1, 64, 32, 45);   // cw1 flat = [45][32]
    for (int tap = 0; tap < 9; ++tap)
        pack_b_f16<<<8, 256, 0, stream>>>(cw2 + tap * 32 * 64, pc2 + tap * 2048, 32, 64, 32);
    pack_b_f16<<<(48 * 8 * 512 + 255) / 256, 256, 0, stream>>>(wx, pwx, 256, 768, 256);
    pack_b_f16<<<(48 * 8 * 512 + 255) / 256, 256, 0, stream>>>(wh, pwh, 256, 768, 256);

    conv1_wmma<<<512, 256, 0, stream>>>(x, pc1, cb1, a1);
    conv2_pool_dense<<<512, 256, 0, stream>>>(a1, pc2, cb2, dw, db, feats);
    xg_gemm<<<192, 256, 0, stream>>>(feats, pwx, bg, xg);
    gru_seq<<<1, 512, 0, stream>>>(xg, pwh, out);
}
